// Top_HiCL_H_9612136808771
// MI455X (gfx1250) — matrix operand order repaired and checked by reading, not compiled, not run on hardware
//
#include <hip/hip_runtime.h>
#include <hip/hip_bf16.h>
#include <math.h>

#define N_NODES 50000
#define EDGES   800000
#define DIM     128
#define PDIM    64
#define KIN     (DIM + PDIM)   // 192
#define DEPTH_  10
#define BATCH   4096
#define KNEG    20
#define TEMP_   0.2f
#define LAMBDA1 1e-5f
#define EPS_    1e-8f

typedef __attribute__((ext_vector_type(16))) _Float16 v16h;
typedef __attribute__((ext_vector_type(4)))  _Float16 v4h;
typedef __attribute__((ext_vector_type(8)))  float    v8f;
typedef __attribute__((ext_vector_type(4)))  unsigned int v4u;

// ---------------------------------------------------------------------------
// Wave32 reduction helper
// ---------------------------------------------------------------------------
__device__ inline float wave_sum(float v) {
#pragma unroll
  for (int off = 16; off > 0; off >>= 1) v += __shfl_xor(v, off, 32);
  return v;
}

// ---------------------------------------------------------------------------
// Build concatenated f16 feature matrix A0[N,192] = [emb_s | emb_p_w[pid]]
// ---------------------------------------------------------------------------
__global__ void build_a0_kernel(const float* __restrict__ emb_s,
                                const float* __restrict__ emb_p_w,
                                const int* __restrict__ pid,
                                _Float16* __restrict__ a0) {
  int i = blockIdx.x * blockDim.x + threadIdx.x;
  if (i >= N_NODES * KIN) return;
  int row = i / KIN;
  int c   = i - row * KIN;
  float v = (c < DIM) ? emb_s[(size_t)row * DIM + c]
                      : emb_p_w[(size_t)pid[row] * PDIM + (c - DIM)];
  a0[i] = (_Float16)v;
}

__global__ void f32_to_f16_kernel(const float* __restrict__ s,
                                  _Float16* __restrict__ d, int n) {
  int i = blockIdx.x * blockDim.x + threadIdx.x;
  if (i < n) d[i] = (_Float16)s[i];
}

// ---------------------------------------------------------------------------
// WMMA GEMM: C[16*grid.x, 128] = act(A[.,K] @ Wt[128,K]^T + bias)
// One 16x16 tile per wave; 8 waves/block cover the full 128 output columns.
// A-frag: lane l holds row (l&15), halves kk+hi*8..+7 and kk+16+hi*8..+7
// B-frag: lane l holds col (l&15) = weight row n, halves kk+hi*8..+7 and kk+16+hi*8..+7
// D elem acc[r] -> row rowBase+r+8*hi, col (l&15).     (ISA 7.12.2, wave32)
// ---------------------------------------------------------------------------
__global__ void gemm_wmma_f16_kernel(const _Float16* __restrict__ A, int K,
                                     const _Float16* __restrict__ Wt,
                                     const float* __restrict__ bias,
                                     float* __restrict__ outF32,
                                     _Float16* __restrict__ outF16,
                                     int relu) {
  const int lane    = threadIdx.x & 31;
  const int wave    = threadIdx.x >> 5;
  const int rowBase = blockIdx.x * 16;
  const int mr      = lane & 15;
  const int hi      = lane >> 4;
  const int n       = wave * 16 + mr;

  const _Float16* Arow = A  + (size_t)(rowBase + mr) * K;
  const _Float16* Wrow = Wt + (size_t)n * K;

  v8f acc = {};
  for (int kk = 0; kk < K; kk += 32) {
    v16h av, bv;
    ((v4u*)&av)[0] = *(const v4u*)(Arow + kk + hi * 8);
    ((v4u*)&av)[1] = *(const v4u*)(Arow + kk + 16 + hi * 8);
    ((v4u*)&bv)[0] = *(const v4u*)(Wrow + kk + hi * 8);
    ((v4u*)&bv)[1] = *(const v4u*)(Wrow + kk + 16 + hi * 8);
    acc = __builtin_amdgcn_wmma_f32_16x16x32_f16(
        /*neg_a=*/false, av, /*neg_b=*/false, bv,
        /*c_mod=*/(short)0, acc, /*reuse_a=*/false, /*reuse_b=*/false);
  }

  const float bn = bias[n];
#pragma unroll
  for (int r = 0; r < 8; ++r) {
    int row = rowBase + r + 8 * hi;
    float v = acc[r] + bn;
    if (relu) v = fmaxf(v, 0.f);
    if (outF32) outF32[(size_t)row * DIM + n] = v;
    if (outF16) outF16[(size_t)row * DIM + n] = (_Float16)v;
  }
}

// ---------------------------------------------------------------------------
// COO sparse: agg[rows[e],:] += vals[e] * h[cols[e],:]   (f16 gather, f32 atomics)
// One wave per edge, grid-stride; prefetch next edge's h row (global_prefetch).
// ---------------------------------------------------------------------------
__global__ void spmm_scatter_kernel(const float* __restrict__ vals,
                                    const int* __restrict__ rows,
                                    const int* __restrict__ cols,
                                    const _Float16* __restrict__ h,
                                    float* __restrict__ agg, int E) {
  const int lane   = threadIdx.x & 31;
  const int nWaves = (gridDim.x * blockDim.x) >> 5;
  int wid = (blockIdx.x * blockDim.x + threadIdx.x) >> 5;
  for (int e = wid; e < E; e += nWaves) {
    if (e + nWaves < E) {
      int cn = cols[e + nWaves];
      __builtin_prefetch(h + (size_t)cn * DIM + lane * 4, 0, 1);
    }
    const float v = vals[e];
    const int c = cols[e];
    const int r = rows[e];
    v4h hv = *(const v4h*)(h + (size_t)c * DIM + lane * 4);
    float* ap = agg + (size_t)r * DIM + lane * 4;
    atomicAdd(ap + 0, v * (float)hv[0]);
    atomicAdd(ap + 1, v * (float)hv[1]);
    atomicAdd(ap + 2, v * (float)hv[2]);
    atomicAdd(ap + 3, v * (float)hv[3]);
  }
}

// ---------------------------------------------------------------------------
// x += agg; refresh f16 mirror of x
// ---------------------------------------------------------------------------
__global__ void residual_kernel(float* __restrict__ x,
                                const float* __restrict__ agg,
                                _Float16* __restrict__ xh, int n) {
  int i = blockIdx.x * blockDim.x + threadIdx.x;
  if (i >= n) return;
  float v = x[i] + agg[i];
  x[i]  = v;
  xh[i] = (_Float16)v;
}

// ---------------------------------------------------------------------------
// InfoNCE: one wave per anchor b. Each lane holds 4 of 128 components.
// accum[0] += sum_k -log(ep/(ep+en_k+EPS))
// ---------------------------------------------------------------------------
__global__ void infonce_kernel(const float* __restrict__ outm,
                               const int* __restrict__ sids,
                               const int* __restrict__ pos,
                               const int* __restrict__ negs,
                               float* __restrict__ accum) {
  const int b    = blockIdx.x;
  const int lane = threadIdx.x;

  const float4 a = *(const float4*)(outm + (size_t)sids[b] * DIM + lane * 4);
  float na2 = wave_sum(a.x * a.x + a.y * a.y + a.z * a.z + a.w * a.w);
  float na  = fmaxf(sqrtf(na2), EPS_);

  auto cos_with = [&](int idx) -> float {
    const float4 q = *(const float4*)(outm + (size_t)idx * DIM + lane * 4);
    float d   = wave_sum(a.x * q.x + a.y * q.y + a.z * q.z + a.w * q.w);
    float nb2 = wave_sum(q.x * q.x + q.y * q.y + q.z * q.z + q.w * q.w);
    float nb  = fmaxf(sqrtf(nb2), EPS_);
    return d / (na * nb);
  };

  float psim = cos_with(pos[b]);
  float ep   = expf(psim / TEMP_);
  float local = 0.f;
  for (int k = 0; k < KNEG; ++k) {
    float ns = cos_with(negs[k * BATCH + b]);
    float en = expf(ns / TEMP_);
    local += -logf(ep / (ep + en + EPS_));
  }
  if (lane == 0) atomicAdd(accum, local);
}

// ---------------------------------------------------------------------------
// sum(p*p) -> accum (atomic)
// ---------------------------------------------------------------------------
__global__ void sumsq_kernel(const float* __restrict__ p, int n,
                             float* __restrict__ accum) {
  float s = 0.f;
  for (int i = blockIdx.x * blockDim.x + threadIdx.x; i < n;
       i += gridDim.x * blockDim.x) {
    float v = p[i];
    s += v * v;
  }
  s = wave_sum(s);
  if ((threadIdx.x & 31) == 0) atomicAdd(accum, s);
}

__global__ void finalize_kernel(const float* __restrict__ accum,
                                float* __restrict__ out) {
  if (threadIdx.x == 0) {
    float lcl  = accum[0] / (float)(BATCH * KNEG);
    float lreg = LAMBDA1 * accum[1];
    out[0] = lcl + lreg;
    out[1] = lcl;
    out[2] = lreg;
  }
}

// ---------------------------------------------------------------------------
extern "C" void kernel_launch(void* const* d_in, const int* in_sizes, int n_in,
                              void* d_out, int out_size, void* d_ws,
                              size_t ws_size, hipStream_t stream) {
  (void)in_sizes; (void)n_in; (void)out_size; (void)ws_size;

  const float* emb_s        = (const float*)d_in[0];
  const float* emb_p_w      = (const float*)d_in[1];
  const float* proj_w       = (const float*)d_in[2];
  const float* proj_b       = (const float*)d_in[3];
  const float* w1           = (const float*)d_in[4];
  const float* b1           = (const float*)d_in[5];
  const float* w2           = (const float*)d_in[6];
  const float* b2           = (const float*)d_in[7];
  const float* out_w        = (const float*)d_in[8];
  const float* out_b        = (const float*)d_in[9];
  const float* adj_vals     = (const float*)d_in[10];
  const int*   adj_rows     = (const int*)d_in[11];
  const int*   adj_cols     = (const int*)d_in[12];
  const int*   position_ids = (const int*)d_in[13];
  const int*   sids         = (const int*)d_in[14];
  const int*   pos          = (const int*)d_in[15];
  const int*   negs         = (const int*)d_in[16];

  char* ws = (char*)d_ws;
  size_t off = 0;
  auto take = [&](size_t bytes) -> char* {
    char* p = ws + off;
    off += (bytes + 255) & ~(size_t)255;
    return p;
  };
  _Float16* a0    = (_Float16*)take((size_t)N_NODES * KIN * 2);
  _Float16* wf    = (_Float16*)take((size_t)(128 * 192 + 3 * 128 * 128) * 2);
  float*    x32   = (float*)take((size_t)N_NODES * DIM * 4);
  _Float16* x16   = (_Float16*)take((size_t)N_NODES * DIM * 2);
  _Float16* h16   = (_Float16*)take((size_t)N_NODES * DIM * 2);
  float*    agg   = (float*)take((size_t)N_NODES * DIM * 4);  // also reused as `out`
  float*    accum = (float*)take(256);

  _Float16* projW16 = wf;
  _Float16* w116    = wf + 128 * 192;
  _Float16* w216    = w116 + 128 * 128;
  _Float16* outw16  = w216 + 128 * 128;

  const int NE = N_NODES * DIM;              // 6.4M
  const int nBlk16 = N_NODES / 16;           // 3125 exact

  // --- feature / weight conversion to f16 ---
  build_a0_kernel<<<(N_NODES * KIN + 255) / 256, 256, 0, stream>>>(
      emb_s, emb_p_w, position_ids, a0);
  f32_to_f16_kernel<<<(128 * 192 + 255) / 256, 256, 0, stream>>>(proj_w, projW16, 128 * 192);
  f32_to_f16_kernel<<<(128 * 128 + 255) / 256, 256, 0, stream>>>(w1, w116, 128 * 128);
  f32_to_f16_kernel<<<(128 * 128 + 255) / 256, 256, 0, stream>>>(w2, w216, 128 * 128);
  f32_to_f16_kernel<<<(128 * 128 + 255) / 256, 256, 0, stream>>>(out_w, outw16, 128 * 128);

  // --- projection: x = A0 @ proj_w.T + proj_b ---
  gemm_wmma_f16_kernel<<<nBlk16, 256, 0, stream>>>(a0, KIN, projW16, proj_b,
                                                   x32, x16, /*relu=*/0);

  // --- GCN layer 1 ---
  gemm_wmma_f16_kernel<<<nBlk16, 256, 0, stream>>>(x16, DIM, w116, b1,
                                                   nullptr, h16, /*relu=*/1);
  hipMemsetAsync(agg, 0, (size_t)NE * 4, stream);
  spmm_scatter_kernel<<<2048, 256, 0, stream>>>(adj_vals, adj_rows, adj_cols,
                                                h16, agg, EDGES);
  residual_kernel<<<(NE + 255) / 256, 256, 0, stream>>>(x32, agg, x16, NE);

  // --- GCN layer 2 ---
  gemm_wmma_f16_kernel<<<nBlk16, 256, 0, stream>>>(x16, DIM, w216, b2,
                                                   nullptr, h16, /*relu=*/1);
  hipMemsetAsync(agg, 0, (size_t)NE * 4, stream);
  spmm_scatter_kernel<<<2048, 256, 0, stream>>>(adj_vals, adj_rows, adj_cols,
                                                h16, agg, EDGES);
  residual_kernel<<<(NE + 255) / 256, 256, 0, stream>>>(x32, agg, x16, NE);

  // --- output layer: out = x @ out_w.T + out_b  (reuse agg as out buffer) ---
  gemm_wmma_f16_kernel<<<nBlk16, 256, 0, stream>>>(x16, DIM, outw16, out_b,
                                                   agg, nullptr, /*relu=*/0);

  // --- losses ---
  hipMemsetAsync(accum, 0, 2 * sizeof(float), stream);
  infonce_kernel<<<BATCH, 32, 0, stream>>>(agg, sids, pos, negs, accum + 0);

  sumsq_kernel<<<4, 256, 0, stream>>>(emb_p_w, DEPTH_ * PDIM, accum + 1);
  sumsq_kernel<<<96, 256, 0, stream>>>(proj_w, 128 * 192, accum + 1);
  sumsq_kernel<<<1, 256, 0, stream>>>(proj_b, 128, accum + 1);
  sumsq_kernel<<<64, 256, 0, stream>>>(w1, 128 * 128, accum + 1);
  sumsq_kernel<<<1, 256, 0, stream>>>(b1, 128, accum + 1);
  sumsq_kernel<<<64, 256, 0, stream>>>(w2, 128 * 128, accum + 1);
  sumsq_kernel<<<1, 256, 0, stream>>>(b2, 128, accum + 1);
  sumsq_kernel<<<64, 256, 0, stream>>>(out_w, 128 * 128, accum + 1);
  sumsq_kernel<<<1, 256, 0, stream>>>(out_b, 128, accum + 1);

  finalize_kernel<<<1, 32, 0, stream>>>(accum, (float*)d_out);
}
